// BidirectionalFG_27032524161628
// MI455X (gfx1250) — matrix operand order repaired and checked by reading, not compiled, not run on hardware
//
#include <hip/hip_runtime.h>

typedef __attribute__((ext_vector_type(16))) _Float16 v16h;
typedef __attribute__((ext_vector_type(8)))  float    v8f;

constexpr int IN_F16_BN  = 0;  // f16 input, apply BN scale/shift + ReLU while staging
constexpr int IN_F32_RAW = 1;  // f32 input, raw (zero-pad K to tile)
constexpr int OUT_F16_STATS = 0;  // f16 output + per-channel sum/sumsq atomics
constexpr int OUT_F32       = 1;  // f32 output (aff / aff_a)
constexpr int OUT_REC       = 2;  // f32 rec output + squared-error reduction

// ---------------------------------------------------------------------------
__global__ void zero_f32_kernel(float* p, int n) {
  for (int i = blockIdx.x * blockDim.x + threadIdx.x; i < n;
       i += gridDim.x * blockDim.x)
    p[i] = 0.f;
}

// First layer: X(M x 3) @ W(3 x 128) + b -> f16 + stats (K too small for WMMA)
__global__ void lin3_kernel(const float* __restrict__ X, int M,
                            const float* __restrict__ W,
                            const float* __restrict__ b,
                            _Float16* __restrict__ Y,
                            float* __restrict__ gSum, float* __restrict__ gSq) {
  __shared__ float rS[256], rQ[256];
  const int c  = threadIdx.x;   // 0..127 (channel)
  const int ty = threadIdx.y;   // 0..1
  const int r0 = blockIdx.x * 128;
  const float w0 = W[c], w1 = W[128 + c], w2 = W[256 + c], bb = b[c];
  float s = 0.f, q = 0.f;
  for (int r = ty; r < 128; r += 2) {
    const int m = r0 + r;
    if (m < M) {
      const float p0 = X[(size_t)m * 3 + 0];
      const float p1 = X[(size_t)m * 3 + 1];
      const float p2 = X[(size_t)m * 3 + 2];
      const float v = fmaf(p0, w0, fmaf(p1, w1, fmaf(p2, w2, bb)));
      Y[(size_t)m * 128 + c] = (_Float16)v;
      s += v; q += v * v;
    }
  }
  rS[ty * 128 + c] = s; rQ[ty * 128 + c] = q;
  __syncthreads();
  if (ty == 0) {
    atomicAdd(&gSum[c], rS[c] + rS[128 + c]);
    atomicAdd(&gSq[c],  rQ[c] + rQ[128 + c]);
  }
}

// Fold batch stats + gamma/beta into per-channel scale/shift
__global__ void bn_finalize_kernel(const float* __restrict__ sum,
                                   const float* __restrict__ sq,
                                   const float* __restrict__ g,
                                   const float* __restrict__ bt, float M,
                                   float* __restrict__ scale,
                                   float* __restrict__ shift) {
  const int c = threadIdx.x;  // 128 threads
  const float mean = sum[c] / M;
  const float var  = fmaxf(sq[c] / M - mean * mean, 0.f);
  const float sc   = g[c] * rsqrtf(var + 1e-5f);
  scale[c] = sc;
  shift[c] = bt[c] - mean * sc;
}

// ---------------------------------------------------------------------------
// WMMA GEMM: block = 128-row tile x NC=NT*16 cols, K = KT (LDS-staged).
// ---------------------------------------------------------------------------
template <int KT, int NT, int MIN, int MOUT>
__global__ __launch_bounds__(256) void gemm_wmma_kernel(
    const void* __restrict__ Xin, int M, int Kreal,
    const float* __restrict__ W, int NCreal,
    const float* __restrict__ bias,
    const float* __restrict__ bnScale, const float* __restrict__ bnShift,
    _Float16* __restrict__ Yf16, float* __restrict__ gSum,
    float* __restrict__ gSq, float* __restrict__ Yf32,
    const float* __restrict__ ref, const int* __restrict__ batch,
    float* __restrict__ segSum, float* __restrict__ segCnt,
    float* __restrict__ epsOut) {
  constexpr int NC = NT * 16;
  constexpr int SA = KT + 8;   // halves; stride*2 bytes is a multiple of 16
  constexpr int SB = NC + 8;
  __shared__ _Float16 ldsA[128 * SA];
  __shared__ _Float16 ldsB[KT * SB];
  __shared__ float redS[128], redQ[128];

  const int tid = threadIdx.x;
  const int r0  = blockIdx.x * 128;

  if constexpr (MOUT == OUT_F16_STATS) {
    if (tid < 128) { redS[tid] = 0.f; redQ[tid] = 0.f; }
  }

  // ---- stage A tile (apply BN+ReLU of previous layer on the fly) ----
  {
    constexpr int CPR  = KT / 8;          // 8-half (16B) chunks per row
    constexpr int ITER = 128 * CPR / 256;
#pragma unroll
    for (int it = 0; it < ITER; ++it) {
      const int idx  = it * 256 + tid;
      const int row  = idx / CPR;
      const int col0 = (idx % CPR) * 8;
      const int m    = r0 + row;
      union { uint4 u; _Float16 h[8]; } o;
      o.u = make_uint4(0u, 0u, 0u, 0u);
      if (m < M) {
        if constexpr (MIN == IN_F16_BN) {
          const _Float16* Xh = (const _Float16*)Xin;
          union { uint4 u; _Float16 h[8]; } in;
          in.u = *(const uint4*)(const void*)(Xh + (size_t)m * KT + col0);
#pragma unroll
          for (int j = 0; j < 8; ++j) {
            const float f =
                fmaf((float)in.h[j], bnScale[col0 + j], bnShift[col0 + j]);
            o.h[j] = (_Float16)fmaxf(f, 0.f);
          }
        } else {
          const float* Xf = (const float*)Xin;
          if (col0 < Kreal) {
            const float4* p =
                (const float4*)(const void*)(Xf + (size_t)m * Kreal + col0);
            const float4 a = p[0], b2 = p[1];
            o.h[0] = (_Float16)a.x;  o.h[1] = (_Float16)a.y;
            o.h[2] = (_Float16)a.z;  o.h[3] = (_Float16)a.w;
            o.h[4] = (_Float16)b2.x; o.h[5] = (_Float16)b2.y;
            o.h[6] = (_Float16)b2.z; o.h[7] = (_Float16)b2.w;
          }
        }
      }
      *(uint4*)(void*)&ldsA[row * SA + col0] = o.u;
    }
  }

  // ---- stage B (weights, f32 -> f16, zero-padded) ----
  {
    constexpr int ITER = KT * NC / 256;
#pragma unroll
    for (int it = 0; it < ITER; ++it) {
      const int idx = it * 256 + tid;
      const int k = idx / NC, n = idx % NC;
      const float v =
          (k < Kreal && n < NCreal) ? W[(size_t)k * NCreal + n] : 0.f;
      ldsB[k * SB + n] = (_Float16)v;
    }
  }

  __syncthreads();

  const int wave = tid >> 5;
  const int lane = tid & 31;
  const int l = lane & 15;
  const int h = lane >> 4;
  const int mloc = wave * 16;   // wave's row base in tile

  v8f acc[NT] = {};
  constexpr int KC = KT / 32;
#pragma unroll
  for (int kc = 0; kc < KC; ++kc) {
    // A 16x32 f16 fragment: lane row = l; halves 0..7 = K[8h..], 8..15 = K[16+8h..]
    union { v16h v; uint4 u[2]; } A;
    const int ab = (mloc + l) * SA + kc * 32 + 8 * h;
    A.u[0] = *(const uint4*)(const void*)&ldsA[ab];
    A.u[1] = *(const uint4*)(const void*)&ldsA[ab + 16];
#pragma unroll
    for (int nt = 0; nt < NT; ++nt) {
      // B 32x16 f16 fragment: lane col = l; halves 0..7 = K[8h..], 8..15 = K[16+8h..]
      union { v16h v; uint4 u[2]; } Bf;
      const int bb = (kc * 32 + 8 * h) * SB + nt * 16 + l;
#pragma unroll
      for (int s = 0; s < 8; ++s) {
        Bf.v[s]     = ldsB[bb + s * SB];
        Bf.v[s + 8] = ldsB[bb + (16 + s) * SB];
      }
      acc[nt] = __builtin_amdgcn_wmma_f32_16x16x32_f16(
          false, A.v, false, Bf.v, (short)0, acc[nt], false, false);
    }
  }

  // ---- epilogue (C/D layout: VGPR r -> row mloc + r + 8h, col = nt*16 + l) ----
  if constexpr (MOUT == OUT_F16_STATS) {
#pragma unroll
    for (int nt = 0; nt < NT; ++nt) {
      const int n = nt * 16 + l;
      const float bb = bias[n];
      float s = 0.f, q = 0.f;
#pragma unroll
      for (int r = 0; r < 8; ++r) {
        const int m = r0 + mloc + r + 8 * h;
        if (m < M) {
          const float v = acc[nt][r] + bb;
          Yf16[(size_t)m * NC + n] = (_Float16)v;
          s += v; q += v * v;
        }
      }
      s += __shfl_xor(s, 16, 32);
      q += __shfl_xor(q, 16, 32);
      if (h == 0) { atomicAdd(&redS[n], s); atomicAdd(&redQ[n], q); }
    }
    __syncthreads();
    if (tid < 128) {
      atomicAdd(&gSum[tid], redS[tid]);
      atomicAdd(&gSq[tid],  redQ[tid]);
    }
  } else if constexpr (MOUT == OUT_F32) {
    const int n = l;
    const float bb = (n < NCreal) ? bias[n] : 0.f;
#pragma unroll
    for (int r = 0; r < 8; ++r) {
      const int m = r0 + mloc + r + 8 * h;
      if (m < M && n < NCreal) Yf32[(size_t)m * NCreal + n] = acc[0][r] + bb;
    }
  } else {  // OUT_REC: write rec + reduce squared error per row
    const int n = l;
    const float bb = (n < NCreal) ? bias[n] : 0.f;
#pragma unroll
    for (int r = 0; r < 8; ++r) {
      const int m = r0 + mloc + r + 8 * h;   // uniform within each 16-lane half
      const bool valid = (m < M);
      const float v = acc[0][r] + bb;
      float contrib = 0.f;
      if (valid && n < NCreal) {
        Yf32[(size_t)m * NCreal + n] = v;
        const float d = ref[(size_t)m * NCreal + n] - v;
        contrib = d * d;
      }
#pragma unroll
      for (int off = 1; off < 16; off <<= 1)
        contrib += __shfl_xor(contrib, off, 32);
      if (valid && n == 0) {
        if (batch) {
          const int bi = batch[m];
          atomicAdd(&segSum[bi], contrib);
          atomicAdd(&segCnt[bi], 1.0f);
        } else {
          epsOut[m] = contrib / (float)NCreal;
        }
      }
    }
  }
}

__global__ void final_kernel(const float* __restrict__ segSum,
                             const float* __restrict__ segCnt,
                             const float* __restrict__ eps,
                             float* __restrict__ eta,
                             float* __restrict__ coher, int Bseg) {
  const int b = blockIdx.x * blockDim.x + threadIdx.x;
  if (b < Bseg) {
    const float c = segCnt[b];
    const float e = (c > 0.f) ? segSum[b] / fmaxf(c, 1.f) : 0.f;
    eta[b] = e;
    coher[b] = e + eps[b];
  }
}

// ---------------------------------------------------------------------------
extern "C" void kernel_launch(void* const* d_in, const int* in_sizes, int n_in,
                              void* d_out, int out_size, void* d_ws,
                              size_t ws_size, hipStream_t stream) {
  const float* pos    = (const float*)d_in[0];
  const int*   batch  = (const int*)d_in[1];
  const float* action = (const float*)d_in[2];
  const int N = in_sizes[0] / 3;
  const int B = in_sizes[2] / 3;

  // Identify MLP blocks (10 arrays each); order of F_inv/G vs sorted-vs-
  // insertion dict flattening resolved via first weight's element count.
  struct MLP { const float *W0,*b0,*g0,*bt0,*W1,*b1,*g1,*bt1,*W2,*b2; };
  MLP F{}, Fi{}, G{}, Gi{};
  int smallSeen = 0, bigSeen = 0;
  for (int blk = 0; blk < 4; ++blk) {
    const int s = 3 + blk * 10;
    if (s + 9 >= n_in) break;
    MLP m;
    m.W0 = (const float*)d_in[s + 0]; m.b0  = (const float*)d_in[s + 1];
    m.g0 = (const float*)d_in[s + 2]; m.bt0 = (const float*)d_in[s + 3];
    m.W1 = (const float*)d_in[s + 4]; m.b1  = (const float*)d_in[s + 5];
    m.g1 = (const float*)d_in[s + 6]; m.bt1 = (const float*)d_in[s + 7];
    m.W2 = (const float*)d_in[s + 8]; m.b2  = (const float*)d_in[s + 9];
    if (in_sizes[s] == 3 * 128) { if (smallSeen++ == 0) F = m; else Fi = m; }
    else                        { if (bigSeen++   == 0) G = m; else Gi = m; }
  }

  // Workspace layout
  float* stats  = (float*)d_ws;          // 8 stages x (sum[128], sq[128])
  float* scsh   = stats + 8 * 256;       // 8 stages x (scale[128], shift[128])
  float* segSum = scsh + 8 * 256;        // B
  float* segCnt = segSum + B;            // B
  _Float16* Xbuf = (_Float16*)(segCnt + B);          // N x 128 f16 (in-place chain)
  _Float16* Ybuf = Xbuf + (size_t)N * 128;           // B x 128 f16
  const size_t need =
      (size_t)(8 * 256 * 2 + 2 * B) * 4 + ((size_t)N + B) * 256;
  if (ws_size < need) return;

  float* out  = (float*)d_out;
  float* eta  = out;
  float* aff  = out + B;
  float* rec  = aff + (size_t)N * 16;
  float* eps  = rec + (size_t)N * 3;
  float* affa = eps + B;
  float* reca = affa + (size_t)B * 16;
  float* coh  = reca + (size_t)B * 3;

  auto sum_ = [&](int st) { return stats + st * 256; };
  auto sq_  = [&](int st) { return stats + st * 256 + 128; };
  auto sc_  = [&](int st) { return scsh + st * 256; };
  auto sh_  = [&](int st) { return scsh + st * 256 + 128; };

  const int zn = 8 * 256 * 2 + 2 * B;
  zero_f32_kernel<<<(zn + 255) / 256, 256, 0, stream>>>(stats, zn);

  const int gp = (N + 127) / 128;
  const int ga = (B + 127) / 128;
  const dim3 b256(256), blin(128, 2);

  // ---------------- point path: pos -> F -> G -> rec, eta ----------------
  lin3_kernel<<<gp, blin, 0, stream>>>(pos, N, F.W0, F.b0, Xbuf, sum_(0), sq_(0));
  bn_finalize_kernel<<<1, 128, 0, stream>>>(sum_(0), sq_(0), F.g0, F.bt0,
                                            (float)N, sc_(0), sh_(0));
  gemm_wmma_kernel<128, 8, IN_F16_BN, OUT_F16_STATS><<<gp, b256, 0, stream>>>(
      Xbuf, N, 128, F.W1, 128, F.b1, sc_(0), sh_(0), Xbuf, sum_(1), sq_(1),
      nullptr, nullptr, nullptr, nullptr, nullptr, nullptr);
  bn_finalize_kernel<<<1, 128, 0, stream>>>(sum_(1), sq_(1), F.g1, F.bt1,
                                            (float)N, sc_(1), sh_(1));
  gemm_wmma_kernel<128, 1, IN_F16_BN, OUT_F32><<<gp, b256, 0, stream>>>(
      Xbuf, N, 128, F.W2, 16, F.b2, sc_(1), sh_(1), nullptr, nullptr, nullptr,
      aff, nullptr, nullptr, nullptr, nullptr, nullptr);
  gemm_wmma_kernel<32, 8, IN_F32_RAW, OUT_F16_STATS><<<gp, b256, 0, stream>>>(
      aff, N, 16, G.W0, 128, G.b0, nullptr, nullptr, Xbuf, sum_(2), sq_(2),
      nullptr, nullptr, nullptr, nullptr, nullptr, nullptr);
  bn_finalize_kernel<<<1, 128, 0, stream>>>(sum_(2), sq_(2), G.g0, G.bt0,
                                            (float)N, sc_(2), sh_(2));
  gemm_wmma_kernel<128, 8, IN_F16_BN, OUT_F16_STATS><<<gp, b256, 0, stream>>>(
      Xbuf, N, 128, G.W1, 128, G.b1, sc_(2), sh_(2), Xbuf, sum_(3), sq_(3),
      nullptr, nullptr, nullptr, nullptr, nullptr, nullptr);
  bn_finalize_kernel<<<1, 128, 0, stream>>>(sum_(3), sq_(3), G.g1, G.bt1,
                                            (float)N, sc_(3), sh_(3));
  gemm_wmma_kernel<128, 1, IN_F16_BN, OUT_REC><<<gp, b256, 0, stream>>>(
      Xbuf, N, 128, G.W2, 3, G.b2, sc_(3), sh_(3), nullptr, nullptr, nullptr,
      rec, pos, batch, segSum, segCnt, nullptr);

  // ------------- action path: action -> F_inv -> G_inv -> epsilon -------------
  lin3_kernel<<<ga, blin, 0, stream>>>(action, B, Fi.W0, Fi.b0, Ybuf, sum_(4), sq_(4));
  bn_finalize_kernel<<<1, 128, 0, stream>>>(sum_(4), sq_(4), Fi.g0, Fi.bt0,
                                            (float)B, sc_(4), sh_(4));
  gemm_wmma_kernel<128, 8, IN_F16_BN, OUT_F16_STATS><<<ga, b256, 0, stream>>>(
      Ybuf, B, 128, Fi.W1, 128, Fi.b1, sc_(4), sh_(4), Ybuf, sum_(5), sq_(5),
      nullptr, nullptr, nullptr, nullptr, nullptr, nullptr);
  bn_finalize_kernel<<<1, 128, 0, stream>>>(sum_(5), sq_(5), Fi.g1, Fi.bt1,
                                            (float)B, sc_(5), sh_(5));
  gemm_wmma_kernel<128, 1, IN_F16_BN, OUT_F32><<<ga, b256, 0, stream>>>(
      Ybuf, B, 128, Fi.W2, 16, Fi.b2, sc_(5), sh_(5), nullptr, nullptr,
      nullptr, affa, nullptr, nullptr, nullptr, nullptr, nullptr);
  gemm_wmma_kernel<32, 8, IN_F32_RAW, OUT_F16_STATS><<<ga, b256, 0, stream>>>(
      affa, B, 16, Gi.W0, 128, Gi.b0, nullptr, nullptr, Ybuf, sum_(6), sq_(6),
      nullptr, nullptr, nullptr, nullptr, nullptr, nullptr);
  bn_finalize_kernel<<<1, 128, 0, stream>>>(sum_(6), sq_(6), Gi.g0, Gi.bt0,
                                            (float)B, sc_(6), sh_(6));
  gemm_wmma_kernel<128, 8, IN_F16_BN, OUT_F16_STATS><<<ga, b256, 0, stream>>>(
      Ybuf, B, 128, Gi.W1, 128, Gi.b1, sc_(6), sh_(6), Ybuf, sum_(7), sq_(7),
      nullptr, nullptr, nullptr, nullptr, nullptr, nullptr);
  bn_finalize_kernel<<<1, 128, 0, stream>>>(sum_(7), sq_(7), Gi.g1, Gi.bt1,
                                            (float)B, sc_(7), sh_(7));
  gemm_wmma_kernel<128, 1, IN_F16_BN, OUT_REC><<<ga, b256, 0, stream>>>(
      Ybuf, B, 128, Gi.W2, 3, Gi.b2, sc_(7), sh_(7), nullptr, nullptr, nullptr,
      reca, action, nullptr, nullptr, nullptr, eps);

  final_kernel<<<(B + 255) / 256, 256, 0, stream>>>(segSum, segCnt, eps, eta,
                                                    coh, B);
}
